// PNN_42167988912745
// MI455X (gfx1250) — matrix operand order repaired and checked by reading, not compiled, not run on hardware
//
#include <hip/hip_runtime.h>
#include <hip/hip_bf16.h>

#define N_FEAT 26
#define VOCAB  100000
#define EMBED  64
#define OUTW   256
#define BATCH  16384
#define K_FLAT (N_FEAT * EMBED)   // 1664
#define M_TILE 16
#define THREADS 256
#define FPAD   32                 // f-dimension padded to 32 (WMMA K)

typedef __attribute__((ext_vector_type(16))) __bf16 v16bf;
typedef __attribute__((ext_vector_type(8)))  float  v8f;

union Frag { unsigned u[8]; v16bf v; };

// ---------------- workspace layout (u32 words) ----------------
// lin_wp : [832][256] u32, word = packed bf16 (k0, k0+1) of lin_w[k][n], k0 = 2*row
#define LIN_WP_WORDS (832 * 256)          // 212992
// inner_wp : [16][256] u32, word = packed bf16 (f0, f0+1) of inner_w[o][f]; rows 13..15 zero
#define INNER_WP_WORDS (16 * 256)         // 4096
// w1p : [256][32] u32, word = packed bf16 (c0, c0+1) of w1[c][j], c0 = 2*row
#define W1P_WORDS (256 * 32)              // 8192

static __device__ inline unsigned pack_bf16(float lo, float hi) {
  union { __bf16 h[2]; unsigned u; } p;
  p.h[0] = (__bf16)lo; p.h[1] = (__bf16)hi;
  return p.u;
}
static __device__ inline unsigned short bf16_bits(float x) {
  union { __bf16 b; unsigned short s; } c; c.b = (__bf16)x; return c.s;
}

// ---------------- prep: pack weights to bf16 pair layout ----------------
__global__ __launch_bounds__(256) void pnn_prep(const float* __restrict__ lin_w,
                                                const float* __restrict__ inner_w,
                                                const float* __restrict__ w1,
                                                unsigned* __restrict__ ws) {
  int t = blockIdx.x * blockDim.x + threadIdx.x;
  if (t < LIN_WP_WORDS) {
    int kp = t >> 8, n = t & 255;
    int k0 = kp * 2;
    ws[t] = pack_bf16(lin_w[(size_t)k0 * OUTW + n], lin_w[(size_t)(k0 + 1) * OUTW + n]);
  } else if (t < LIN_WP_WORDS + INNER_WP_WORDS) {
    int t2 = t - LIN_WP_WORDS;
    int kp = t2 >> 8, o = t2 & 255;
    int f0 = kp * 2;
    unsigned v = 0u;
    if (f0 < N_FEAT)     // kp <= 12: f0 <= 24, f0+1 <= 25 both valid; kp 13..15 -> 0
      v = pack_bf16(inner_w[o * N_FEAT + f0], inner_w[o * N_FEAT + f0 + 1]);
    ws[LIN_WP_WORDS + t2] = v;
  } else if (t < LIN_WP_WORDS + INNER_WP_WORDS + W1P_WORDS) {
    int t3 = t - LIN_WP_WORDS - INNER_WP_WORDS;
    int cp = t3 >> 5, j = t3 & 31;
    int c0 = cp * 2;
    ws[LIN_WP_WORDS + INNER_WP_WORDS + t3] =
        pack_bf16(w1[c0 * 32 + j], w1[(c0 + 1) * 32 + j]);
  }
}

// ---------------- main fused kernel ----------------
// LDS layout (bytes):
#define OFF_EMBA 0
#define SZ_EMBA  (M_TILE * K_FLAT * 2)            // 53248: bf16 [m][f*64+e]
#define OFF_EMBT (OFF_EMBA + SZ_EMBA)
#define SZ_EMBT  (M_TILE * EMBED * FPAD * 2)      // 65536: bf16 [m][e][f(pad32)]
#define OFF_IDX  (OFF_EMBT + SZ_EMBT)
#define SZ_IDX   (M_TILE * N_FEAT * 4)            // 1664
#define SMEM_BYTES (OFF_IDX + SZ_IDX)             // 120448
// epilogue buffers overlap EMBA region (only touched after a barrier):
#define OFF_PROD 0                                 // bf16 [16][512] = 16384
#define OFF_H    (OFF_PROD + M_TILE * 512 * 2)     // f32  [16][32]  = 2048

__global__ __launch_bounds__(THREADS) void pnn_main(
    const int*   __restrict__ inputs,
    const float* __restrict__ tables,
    const float* __restrict__ lin_b,
    const float* __restrict__ b1,
    const float* __restrict__ w2, const float* __restrict__ b2,
    const unsigned* __restrict__ lin_wp,
    const unsigned* __restrict__ inner_wp,
    const unsigned* __restrict__ w1p,
    float* __restrict__ out)
{
  __shared__ __align__(16) char smem[SMEM_BYTES];
  const int t    = threadIdx.x;
  const int lane = t & 31;
  const int wave = t >> 5;
  const int half = lane >> 4;          // which 16-lane half
  const int lan  = lane & 15;
  const int row0 = blockIdx.x * M_TILE;

  int*            s_idx  = (int*)(smem + OFF_IDX);
  unsigned short* embA16 = (unsigned short*)(smem + OFF_EMBA);
  unsigned short* embT16 = (unsigned short*)(smem + OFF_EMBT);
  unsigned*       embTw  = (unsigned*)(smem + OFF_EMBT);
  const unsigned* embA32 = (const unsigned*)(smem + OFF_EMBA);
  const unsigned* embT32 = (const unsigned*)(smem + OFF_EMBT);

  // ---- stage 0: indices + zero the embT f-padding (f = 26..31) ----
  for (int i = t; i < M_TILE * N_FEAT; i += THREADS)
    s_idx[i] = inputs[row0 * N_FEAT + i];
  #pragma unroll
  for (int p = 0; p < 4; ++p) {
    int pe = t + p * THREADS;                 // 0..1023 = m*64 + e
    int m  = pe >> 6, e = pe & 63;
    int w  = m * (EMBED * FPAD / 2) + e * (FPAD / 2) + (N_FEAT / 2);
    embTw[w] = 0u; embTw[w + 1] = 0u; embTw[w + 2] = 0u;
  }
  __syncthreads();

  // ---- stage 1: gather embeddings, f32 -> bf16, two LDS layouts ----
  for (int iter = 0; iter < 13; ++iter) {
    int task = t + iter * THREADS;
    int pair = task >> 3;
    int e0   = (task & 7) * 8;
    int m    = pair / N_FEAT;
    int f    = pair - m * N_FEAT;
    int idx  = s_idx[pair];
    const float* src = tables + ((size_t)f * VOCAB + (size_t)idx) * EMBED + e0;
    if (iter < 12) {
      int taskN = task + THREADS;
      int pairN = taskN >> 3;
      int e0N   = (taskN & 7) * 8;
      int mN    = pairN / N_FEAT;
      int fN    = pairN - mN * N_FEAT;
      __builtin_prefetch(tables + ((size_t)fN * VOCAB + (size_t)s_idx[pairN]) * EMBED + e0N, 0, 0);
    }
    float4 x0 = *(const float4*)(src);
    float4 x1 = *(const float4*)(src + 4);
    union { unsigned u[4]; uint4 q; } pk;
    pk.u[0] = pack_bf16(x0.x, x0.y);
    pk.u[1] = pack_bf16(x0.z, x0.w);
    pk.u[2] = pack_bf16(x1.x, x1.y);
    pk.u[3] = pack_bf16(x1.z, x1.w);
    *(uint4*)(embA16 + (size_t)m * K_FLAT + f * EMBED + e0) = pk.q;
    unsigned short hb[8] = { bf16_bits(x0.x), bf16_bits(x0.y), bf16_bits(x0.z), bf16_bits(x0.w),
                             bf16_bits(x1.x), bf16_bits(x1.y), bf16_bits(x1.z), bf16_bits(x1.w) };
    #pragma unroll
    for (int ee = 0; ee < 8; ++ee)
      embT16[(size_t)m * (EMBED * FPAD) + (e0 + ee) * FPAD + f] = hb[ee];
  }
  __syncthreads();

  // Each wave owns two 16-wide N tiles.
  const int ncol0 = wave * 32 + lan;
  const int ncol1 = wave * 32 + 16 + lan;

  // ---- GEMM1: lz = flat @ lin_w   (K = 1664 = 52 * 32) ----
  v8f acc0 = {}, acc1 = {};
  for (int kb = 0; kb < 52; ++kb) {
    Frag a;
    #pragma unroll
    for (int j = 0; j < 8; ++j) {
      int k0 = kb * 32 + 2 * j + 16 * half;  // even
      a.u[j] = embA32[(lan * K_FLAT + k0) >> 1];
    }
    const int kprow = kb * 16 + 8 * half;
    Frag b0, b1f;
    #pragma unroll
    for (int j = 0; j < 8; ++j) {
      b0.u[j]  = lin_wp[(kprow + j) * OUTW + ncol0];
      b1f.u[j] = lin_wp[(kprow + j) * OUTW + ncol1];
    }
    acc0 = __builtin_amdgcn_wmma_f32_16x16x32_bf16(false, a.v, false, b0.v, (short)0, acc0, false, false);
    acc1 = __builtin_amdgcn_wmma_f32_16x16x32_bf16(false, a.v, false, b1f.v, (short)0, acc1, false, false);
  }

  // ---- GEMM2: delta_e = emb[:, :, e] @ inner_w^T, lp += delta_e^2 ----
  Frag wb0, wb1;
  #pragma unroll
  for (int j = 0; j < 8; ++j) {
    wb0.u[j] = inner_wp[(8 * half + j) * OUTW + ncol0];
    wb1.u[j] = inner_wp[(8 * half + j) * OUTW + ncol1];
  }
  v8f lp0 = {}, lp1 = {};
  for (int e = 0; e < EMBED; ++e) {
    Frag a;
    #pragma unroll
    for (int j = 0; j < 8; ++j) {
      int k0 = 2 * j + 16 * half;            // even, pad -> 0
      a.u[j] = embT32[(lan * (EMBED * FPAD) + e * FPAD + k0) >> 1];
    }
    v8f z = {};
    v8f d0 = __builtin_amdgcn_wmma_f32_16x16x32_bf16(false, a.v, false, wb0.v, (short)0, z, false, false);
    v8f d1 = __builtin_amdgcn_wmma_f32_16x16x32_bf16(false, a.v, false, wb1.v, (short)0, z, false, false);
    #pragma unroll
    for (int v = 0; v < 8; ++v) {
      lp0[v] = fmaf(d0[v], d0[v], lp0[v]);
      lp1[v] = fmaf(d1[v], d1[v], lp1[v]);
    }
  }

  // ---- epilogue: prod = [lz + bias | sqrt(lp)] as bf16 into LDS (overlaps embA) ----
  __syncthreads();   // all waves finished reading emb LDS
  unsigned short* prodb = (unsigned short*)(smem + OFF_PROD);
  const unsigned* prodw = (const unsigned*)(smem + OFF_PROD);
  float*          s_h   = (float*)(smem + OFF_H);
  const float bias0 = lin_b[ncol0];
  const float bias1 = lin_b[ncol1];
  #pragma unroll
  for (int v = 0; v < 8; ++v) {
    int m = v + 8 * half;                // C/D layout: lane<16 -> M=v, lane>=16 -> M=v+8
    prodb[m * 512 + ncol0]       = bf16_bits(acc0[v] + bias0);
    prodb[m * 512 + ncol1]       = bf16_bits(acc1[v] + bias1);
    prodb[m * 512 + 256 + ncol0] = bf16_bits(sqrtf(lp0[v]));
    prodb[m * 512 + 256 + ncol1] = bf16_bits(sqrtf(lp1[v]));
  }
  // init h accumulator with bias b1 (relu applied at final read)
  s_h[t]           = b1[t & 31];
  s_h[t + THREADS] = b1[t & 31];
  __syncthreads();

  // ---- MLP layer 1 via WMMA: h(16x32) = prod(16x512) @ w1(512x32) ----
  // split-K: wave w handles K-steps {2w, 2w+1}; partials merged with ds_add_f32
  {
    v8f h0 = {}, h1 = {};
    #pragma unroll
    for (int s = 0; s < 2; ++s) {
      const int kb = wave * 2 + s;
      Frag a;
      #pragma unroll
      for (int j = 0; j < 8; ++j) {
        int koff = 2 * j + 16 * half;
        a.u[j] = prodw[(lan * 512 + kb * 32 + koff) >> 1];
      }
      const int kprow = kb * 16 + 8 * half;
      Frag bw0, bw1;
      #pragma unroll
      for (int j = 0; j < 8; ++j) {
        bw0.u[j] = w1p[(kprow + j) * 32 + lan];
        bw1.u[j] = w1p[(kprow + j) * 32 + 16 + lan];
      }
      h0 = __builtin_amdgcn_wmma_f32_16x16x32_bf16(false, a.v, false, bw0.v, (short)0, h0, false, false);
      h1 = __builtin_amdgcn_wmma_f32_16x16x32_bf16(false, a.v, false, bw1.v, (short)0, h1, false, false);
    }
    #pragma unroll
    for (int v = 0; v < 8; ++v) {
      int m = v + 8 * half;
      atomicAdd(&s_h[m * 32 + lan],      h0[v]);
      atomicAdd(&s_h[m * 32 + 16 + lan], h1[v]);
    }
  }
  __syncthreads();

  // ---- out = sigmoid(relu(h) @ w2 + b2) ----
  if (t < M_TILE) {
    float acc = b2[0];
    #pragma unroll
    for (int j = 0; j < 32; ++j) acc = fmaf(fmaxf(s_h[t * 32 + j], 0.0f), w2[j], acc);
    out[row0 + t] = 1.0f / (1.0f + expf(-acc));
  }
}

// ---------------- host launcher ----------------
extern "C" void kernel_launch(void* const* d_in, const int* in_sizes, int n_in,
                              void* d_out, int out_size, void* d_ws, size_t ws_size,
                              hipStream_t stream) {
  const int*   inputs  = (const int*)d_in[0];
  const float* tables  = (const float*)d_in[1];
  const float* lin_w   = (const float*)d_in[2];
  const float* lin_b   = (const float*)d_in[3];
  const float* inner_w = (const float*)d_in[4];
  const float* w1      = (const float*)d_in[5];
  const float* b1      = (const float*)d_in[6];
  const float* w2      = (const float*)d_in[7];
  const float* b2      = (const float*)d_in[8];
  float* out = (float*)d_out;

  unsigned* ws       = (unsigned*)d_ws;
  unsigned* lin_wp   = ws;
  unsigned* inner_wp = ws + LIN_WP_WORDS;
  unsigned* w1p      = ws + LIN_WP_WORDS + INNER_WP_WORDS;

  const int prep_total = LIN_WP_WORDS + INNER_WP_WORDS + W1P_WORDS;
  pnn_prep<<<(prep_total + 255) / 256, 256, 0, stream>>>(lin_w, inner_w, w1, ws);
  pnn_main<<<BATCH / M_TILE, THREADS, 0, stream>>>(inputs, tables, lin_b, b1, w2, b2,
                                                   lin_wp, inner_wp, w1p, out);
}
